// ERC_EFR_multitask_50749333569964
// MI455X (gfx1250) — matrix operand order repaired and checked by reading, not compiled, not run on hardware
//
#include <hip/hip_runtime.h>
#include <cstdint>

#define BATCH 8
#define SEQ   1024
#define HDIM  768

// ---- CDNA5 WMMA fragment types (probe-verified builtin signature) ----
typedef __attribute__((ext_vector_type(16))) __bf16 v16bf;
typedef __attribute__((ext_vector_type(8)))  float  v8f;
typedef __attribute__((ext_vector_type(4)))  unsigned su4;
typedef __attribute__((ext_vector_type(8)))  unsigned su8;

union ABfrag { v16bf v; uint4 q[2]; };

__device__ __forceinline__ unsigned short f2bf(float f) {
  union { float f; unsigned u; } x; x.f = f;
  unsigned r = x.u + 0x7FFFu + ((x.u >> 16) & 1u);   // RNE
  return (unsigned short)(r >> 16);
}
__device__ __forceinline__ float sigmoidf_(float x) { return 1.0f / (1.0f + __expf(-x)); }
__device__ __forceinline__ float tanhf_(float x)    { float e = __expf(2.0f * x); return (e - 1.0f) / (e + 1.0f); }

// LDS byte offset of a __shared__ object: generic LDS address low 32 bits
// are the offset within the workgroup's LDS allocation (aperture mapping).
__device__ __forceinline__ unsigned lds_off(const void* p) {
  return (unsigned)(uintptr_t)p;
}

// ------------------------------------------------------------------
// Tensor Data Mover: async 2-D tile load global -> LDS.
// D# group0 (4 SGPRs): count=1 | lds_addr | global_addr | type=2
// D# group1 (8 SGPRs): data_size=2B, tile 32 x rows, row stride = K elems.
// 2-operand form => VADDR2/VADDR3 NULL => tensor up to 2D.
// Completion tracked with TENSORcnt (s_wait_tensorcnt).
// ------------------------------------------------------------------
__device__ __forceinline__ void tdm_load_tile_b16(
    unsigned lds_byte_off, const unsigned short* gptr,
    unsigned tile_rows, unsigned row_stride_elems)
{
  unsigned long long ga = (unsigned long long)(uintptr_t)gptr;
  su4 g0;
  g0[0] = 1u;                                    // count=1, user descriptor
  g0[1] = lds_byte_off;                          // lds_addr (bytes)
  g0[2] = (unsigned)ga;                          // global_addr[31:0]
  g0[3] = ((unsigned)(ga >> 32) & 0x01FFFFFFu)   // global_addr[56:32]
          | (2u << 30);                          // type = 2 ("image")
  su8 g1;
  g1[0] = 1u << 16;                              // wg_mask=0, data_size=1 (2B)
  g1[1] = 0xFFFFu << 16;                         // tensor_dim0[15:0]
  g1[2] = 0x7FFFu | (0xFFFFu << 16);             // tensor_dim0[31:16], tensor_dim1[15:0]
  g1[3] = 0x7FFFu | (32u << 16);                 // tensor_dim1[31:16], tile_dim0=32
  g1[4] = tile_rows & 0xFFFFu;                   // tile_dim1=rows, tile_dim2=0
  g1[5] = row_stride_elems;                      // tensor_dim0_stride[31:0]
  g1[6] = 0u;                                    // stride[47:32]=0, dim1_stride[15:0]=0
  g1[7] = 0u;                                    // dim1_stride[47:16]=0
  asm volatile("tensor_load_to_lds %0, %1" :: "s"(g0), "s"(g1) : "memory");
}

// ------------------------------------------------------------------
// Utility kernels
// ------------------------------------------------------------------
__global__ void zero_u32_kernel(unsigned* __restrict__ p, int n) {
  for (int i = blockIdx.x * blockDim.x + threadIdx.x; i < n; i += gridDim.x * blockDim.x)
    p[i] = 0u;
}

__global__ void cvt_f32_bf16_kernel(const float* __restrict__ s, unsigned short* __restrict__ d, int n) {
  for (int i = blockIdx.x * blockDim.x + threadIdx.x; i < n; i += gridDim.x * blockDim.x)
    d[i] = f2bf(s[i]);
}

// ------------------------------------------------------------------
// Big parallel GEMM:  Y[M,N] = Xbf16[M,K] @ Wbf16[N,K]^T + bias[N]
// 128x128 workgroup tile, 8 waves, each wave 32x64 (2x4 wmma tiles).
// Tiles staged by the Tensor Data Mover into double-buffered LDS:
// wave0 issues TDM for buffer (b^1) while all waves compute from b.
// A-frag: lane l<16 -> row l, K chunks {0..7},{16..23}; l>=16 -> +8.
// B-frag: lane l<16 -> col l, K chunks {0..7},{16..23}; l>=16 -> +8.
// C/D:    lanes 0..15 -> M = vgpr 0..7; lanes 16..31 -> M = 8..15.
// ------------------------------------------------------------------
__global__ __launch_bounds__(256) void wmma_gemm_xg(
    const unsigned short* __restrict__ X,   // [M,K] bf16
    const unsigned short* __restrict__ W,   // [N,K] bf16 (row-major = B^T)
    const float* __restrict__ bias,         // [N]
    float* __restrict__ Y,                  // [M,N] f32
    int M, int N, int K)
{
  (void)M;
  __shared__ __attribute__((aligned(16))) unsigned short lA[2][128 * 32];
  __shared__ __attribute__((aligned(16))) unsigned short lB[2][128 * 32];
  const int tid  = threadIdx.x;
  const int wave = tid >> 5, lane = tid & 31;
  const int half = lane >> 4, lr = lane & 15;
  const int wm = (wave & 3) * 32;       // 4 waves along M
  const int wn = (wave >> 2) * 64;      // 2 waves along N
  const int rowA0 = blockIdx.y * 128;
  const int colB0 = blockIdx.x * 128;
  const bool issuer = (wave == 0);

  const unsigned offA[2] = { lds_off(&lA[0][0]), lds_off(&lA[1][0]) };
  const unsigned offB[2] = { lds_off(&lB[0][0]), lds_off(&lB[1][0]) };
  const unsigned short* Xt = &X[(size_t)rowA0 * K];
  const unsigned short* Wt = &W[(size_t)colB0 * K];

  v8f acc[2][4] = {};

  // Prime buffer 0 with the k=0 tiles.
  if (issuer) {
    tdm_load_tile_b16(offA[0], Xt, 128, (unsigned)K);
    tdm_load_tile_b16(offB[0], Wt, 128, (unsigned)K);
    __builtin_amdgcn_s_wait_tensorcnt(0);
  }
  __syncthreads();

  int buf = 0;
  for (int kt = 0; kt < K; kt += 32) {
    // Prefetch next K-tile into the other buffer via TDM (overlaps WMMA).
    if (issuer && (kt + 32) < K) {
      tdm_load_tile_b16(offA[buf ^ 1], Xt + (kt + 32), 128, (unsigned)K);
      tdm_load_tile_b16(offB[buf ^ 1], Wt + (kt + 32), 128, (unsigned)K);
    }

    ABfrag a[2], b[4];
#pragma unroll
    for (int mi = 0; mi < 2; ++mi) {
      const unsigned short* p = &lA[buf][(wm + mi * 16 + lr) * 32];
      a[mi].q[0] = *(const uint4*)&p[half * 8];
      a[mi].q[1] = *(const uint4*)&p[16 + half * 8];
    }
#pragma unroll
    for (int ni = 0; ni < 4; ++ni) {
      const unsigned short* p = &lB[buf][(wn + ni * 16 + lr) * 32 + half * 8];
      b[ni].q[0] = *(const uint4*)&p[0];
      b[ni].q[1] = *(const uint4*)&p[16];
    }
#pragma unroll
    for (int mi = 0; mi < 2; ++mi)
#pragma unroll
      for (int ni = 0; ni < 4; ++ni)
        acc[mi][ni] = __builtin_amdgcn_wmma_f32_16x16x32_bf16(
            false, a[mi].v, false, b[ni].v, (short)0, acc[mi][ni], false, false);

    if (issuer) __builtin_amdgcn_s_wait_tensorcnt(0);  // next-tile DMA done
    __syncthreads();
    buf ^= 1;
  }

#pragma unroll
  for (int mi = 0; mi < 2; ++mi) {
#pragma unroll
    for (int ni = 0; ni < 4; ++ni) {
      int col = colB0 + wn + ni * 16 + lr;
      float bv = bias[col];
#pragma unroll
      for (int r = 0; r < 8; ++r) {
        int row = rowA0 + wm + mi * 16 + half * 8 + r;
        Y[(size_t)row * N + col] = acc[mi][ni][r] + bv;
      }
    }
  }
}

// ------------------------------------------------------------------
// Persistent GRU recurrence. One block per 16 hidden outputs.
// Per step: gh = h @ W_hh^T for this block's 3 gate tiles (M padded
// 8->16, K split across 4 waves, bf16 WMMA, LDS reduce), then gate
// math on wave0 lanes 0..15, then a grid-wide release/acquire barrier.
// hB: [16, Hh] bf16 state (rows 8..15 stay zero). hF: [8, Hh] f32.
// ------------------------------------------------------------------
__global__ __launch_bounds__(128) void gru_recur_kernel(
    const unsigned short* __restrict__ Whh,  // [3*Hh, Hh] bf16
    const float* __restrict__ xg,            // [BATCH*SEQ, 3*Hh] (includes b_ih)
    const float* __restrict__ bhh,           // [3*Hh]
    float* __restrict__ out,                 // [BATCH*SEQ, Hh] f32
    unsigned short* __restrict__ out_b,      // optional bf16 copy (feeds next GEMMs)
    float* __restrict__ hF,                  // [BATCH, Hh] f32 state
    unsigned short* __restrict__ hB,         // [16, Hh] bf16 state (padded)
    unsigned* __restrict__ cnt,              // grid barrier counter (zeroed)
    int Hh, int nblocks)
{
  const int hb   = blockIdx.x * 16;
  const int tid  = threadIdx.x;
  const int wave = tid >> 5, lane = tid & 31;
  const int half = lane >> 4, lr = lane & 15;
  const int KTW  = (Hh >> 5) >> 2;      // K tiles per wave (12 or 6)
  const int N3   = 3 * Hh;
  __shared__ __attribute__((aligned(32))) float red[4 * 3 * 256];

  for (int t = 0; t < SEQ; ++t) {
    v8f acc[3] = {};
    for (int q = 0; q < KTW; ++q) {
      int kt = (wave * KTW + q) << 5;
      ABfrag a;
      const unsigned short* pa = &hB[(size_t)lr * Hh + kt + half * 8];
      a.q[0] = *(const uint4*)pa;
      a.q[1] = *(const uint4*)(pa + 16);
#pragma unroll
      for (int g = 0; g < 3; ++g) {
        ABfrag b;
        const unsigned short* pb = &Whh[(size_t)(g * Hh + hb + lr) * Hh + kt + half * 8];
        b.q[0] = *(const uint4*)pb;
        b.q[1] = *(const uint4*)(pb + 16);
        acc[g] = __builtin_amdgcn_wmma_f32_16x16x32_bf16(
            false, a.v, false, b.v, (short)0, acc[g], false, false);
      }
    }
#pragma unroll
    for (int g = 0; g < 3; ++g)
      *(v8f*)&red[((wave * 3 + g) << 8) + (lane << 3)] = acc[g];
    __syncthreads();

    if (wave == 0 && half == 0) {       // lanes 0..15: C rows M=0..7 (real batch)
      v8f sg[3];
#pragma unroll
      for (int g = 0; g < 3; ++g) {
        v8f s = *(const v8f*)&red[((0 * 3 + g) << 8) + (lane << 3)];
        s = s + *(const v8f*)&red[((1 * 3 + g) << 8) + (lane << 3)];
        s = s + *(const v8f*)&red[((2 * 3 + g) << 8) + (lane << 3)];
        s = s + *(const v8f*)&red[((3 * 3 + g) << 8) + (lane << 3)];
        sg[g] = s;
      }
      const int j = hb + lr;
      const float bh_r = bhh[j], bh_z = bhh[Hh + j], bh_n = bhh[2 * Hh + j];
#pragma unroll
      for (int m = 0; m < 8; ++m) {
        const float* xrow = &xg[(size_t)(m * SEQ + t) * N3];
        float rr = sigmoidf_(xrow[j]          + sg[0][m] + bh_r);
        float zz = sigmoidf_(xrow[Hh + j]     + sg[1][m] + bh_z);
        float nn = tanhf_   (xrow[2 * Hh + j] + rr * (sg[2][m] + bh_n));
        float hp = hF[(size_t)m * Hh + j];
        float hv = (1.0f - zz) * nn + zz * hp;
        hF[(size_t)m * Hh + j] = hv;
        hB[(size_t)m * Hh + j] = f2bf(hv);
        out[(size_t)(m * SEQ + t) * Hh + j] = hv;
        if (out_b) out_b[(size_t)(m * SEQ + t) * Hh + j] = f2bf(hv);
      }
      __threadfence();                  // release this step's hB/hF writes
    }
    __syncthreads();

    if (t + 1 < SEQ) {                  // grid-wide barrier
      if (tid == 0) {
        atomicAdd(cnt, 1u);
        unsigned target = (unsigned)nblocks * (unsigned)(t + 1);
        while (__hip_atomic_load(cnt, __ATOMIC_RELAXED, __HIP_MEMORY_SCOPE_AGENT) < target)
          __builtin_amdgcn_s_sleep(2);
      }
      __syncthreads();
      __threadfence();                  // acquire: drop stale hB lines before reload
    }
  }
}

// ------------------------------------------------------------------
// fop1[row, 0..6] = ((rnn_e + rnn_c)[row] . W1[g]) + b1[g]
// ------------------------------------------------------------------
__global__ __launch_bounds__(256) void fop1_kernel(
    const float* __restrict__ e, const float* __restrict__ c,
    const float* __restrict__ W1, const float* __restrict__ b1,
    float* __restrict__ out)
{
  const int row = blockIdx.x;
  const int tid = threadIdx.x;
  __shared__ float red[7 * 256];
  float s[7] = {};
  const float* pe = &e[(size_t)row * 1536];
  const float* pc = &c[(size_t)row * 1536];
  for (int k = tid; k < 1536; k += 256) {
    float v = pe[k] + pc[k];
#pragma unroll
    for (int g = 0; g < 7; ++g) s[g] += v * W1[g * 1536 + k];
  }
#pragma unroll
  for (int g = 0; g < 7; ++g) red[g * 256 + tid] = s[g];
  __syncthreads();
  for (int off = 128; off > 0; off >>= 1) {
    if (tid < off)
#pragma unroll
      for (int g = 0; g < 7; ++g) red[g * 256 + tid] += red[g * 256 + tid + off];
    __syncthreads();
  }
  if (tid < 7) out[(size_t)row * 7 + tid] = red[tid * 256] + b1[tid];
}

// ------------------------------------------------------------------
// AB[row] = {A0, A1, Bv0, Bv1}:  A = rnn_t . W2[:, :768]^T, Bv = rnn_t . W2[:, 768:]^T
// ------------------------------------------------------------------
__global__ __launch_bounds__(256) void proj2_kernel(
    const float* __restrict__ tt, const float* __restrict__ W2,
    float* __restrict__ AB)
{
  const int row = blockIdx.x;
  const int tid = threadIdx.x;
  __shared__ float red[4 * 256];
  float s0 = 0.f, s1 = 0.f, s2 = 0.f, s3 = 0.f;
  const float* p = &tt[(size_t)row * HDIM];
  for (int k = tid; k < HDIM; k += 256) {
    float v = p[k];
    s0 += v * W2[k];                 // W2[0][k]
    s1 += v * W2[1536 + k];          // W2[1][k]
    s2 += v * W2[768 + k];           // W2[0][768+k]
    s3 += v * W2[1536 + 768 + k];    // W2[1][768+k]
  }
  red[tid] = s0; red[256 + tid] = s1; red[512 + tid] = s2; red[768 + tid] = s3;
  __syncthreads();
  for (int off = 128; off > 0; off >>= 1) {
    if (tid < off) {
      red[tid]       += red[tid + off];
      red[256 + tid] += red[256 + tid + off];
      red[512 + tid] += red[512 + tid + off];
      red[768 + tid] += red[768 + tid + off];
    }
    __syncthreads();
  }
  if (tid < 4) AB[(size_t)row * 4 + tid] = red[tid * 256];
}

// ------------------------------------------------------------------
// fop2[b, s, w, c] = A[b,s,c] + Bv[b, s-4+w, c] + b2[c]  (0 where s-4+w < 0)
// ------------------------------------------------------------------
__global__ void fop2_kernel(const float* __restrict__ AB,
                            const float* __restrict__ b2,
                            float* __restrict__ out)
{
  int i = blockIdx.x * blockDim.x + threadIdx.x;
  if (i >= BATCH * SEQ * 5 * 2) return;
  int c  = i & 1;
  int w  = (i >> 1) % 5;
  int s  = (i / 10) % SEQ;
  int bb = i / (10 * SEQ);
  int raw = s - 4 + w;
  float v = 0.0f;
  if (raw >= 0)
    v = AB[(size_t)(bb * SEQ + s) * 4 + c] + AB[(size_t)(bb * SEQ + raw) * 4 + 2 + c] + b2[c];
  out[i] = v;
}

// ------------------------------------------------------------------
extern "C" void kernel_launch(void* const* d_in, const int* in_sizes, int n_in,
                              void* d_out, int out_size, void* d_ws, size_t ws_size,
                              hipStream_t stream)
{
  (void)in_sizes; (void)n_in; (void)out_size; (void)ws_size;
  const float* op     = (const float*)d_in[0];
  const float* W_ih_c = (const float*)d_in[1];
  const float* W_hh_c = (const float*)d_in[2];
  const float* b_ih_c = (const float*)d_in[3];
  const float* b_hh_c = (const float*)d_in[4];
  const float* W_ih_e = (const float*)d_in[5];
  const float* W_hh_e = (const float*)d_in[6];
  const float* b_ih_e = (const float*)d_in[7];
  const float* b_hh_e = (const float*)d_in[8];
  const float* W_ih_t = (const float*)d_in[9];
  const float* W_hh_t = (const float*)d_in[10];
  const float* b_ih_t = (const float*)d_in[11];
  const float* b_hh_t = (const float*)d_in[12];
  const float* W1     = (const float*)d_in[13];
  const float* b1     = (const float*)d_in[14];
  const float* W2     = (const float*)d_in[15];
  const float* b2     = (const float*)d_in[16];
  float* out = (float*)d_out;

  const int MR = BATCH * SEQ;   // 8192 rows

  char* wp = (char*)d_ws;
  auto alloc = [&](size_t bytes) -> char* {
    char* p = wp; wp += (bytes + 255) & ~(size_t)255; return p;
  };
  unsigned short* opb    = (unsigned short*)alloc((size_t)MR * 2304 * 2);
  unsigned short* Wihcb  = (unsigned short*)alloc((size_t)4608 * 2304 * 2);
  unsigned short* Whhcb  = (unsigned short*)alloc((size_t)4608 * 1536 * 2);
  unsigned short* Wiheb  = (unsigned short*)alloc((size_t)4608 * 1536 * 2);
  unsigned short* Whheb  = (unsigned short*)alloc((size_t)4608 * 1536 * 2);
  unsigned short* Wihtb  = (unsigned short*)alloc((size_t)2304 * 1536 * 2);
  unsigned short* Whhtb  = (unsigned short*)alloc((size_t)2304 * 768 * 2);
  float*          xg_c   = (float*)alloc((size_t)MR * 4608 * 4);
  float*          xg_t   = (float*)alloc((size_t)MR * 2304 * 4);
  float*          rnn_c  = (float*)alloc((size_t)MR * 1536 * 4);
  unsigned short* rnn_cb = (unsigned short*)alloc((size_t)MR * 1536 * 2);
  float*          rnn_e  = (float*)alloc((size_t)MR * 1536 * 4);
  float*          rnn_t  = (float*)alloc((size_t)MR * 768 * 4);
  float*          ABbuf  = (float*)alloc((size_t)MR * 4 * 4);
  float*          hF     = (float*)alloc((size_t)8 * 1536 * 4);
  unsigned short* hB     = (unsigned short*)alloc((size_t)16 * 1536 * 2);
  unsigned*       cnts   = (unsigned*)alloc(256);
  float*          xg_e   = xg_c;   // xg_c is dead once recurrence-c finishes

  // --- f32 -> bf16 conversions ---
  cvt_f32_bf16_kernel<<<2048, 256, 0, stream>>>(op,     opb,   MR * 2304);
  cvt_f32_bf16_kernel<<<2048, 256, 0, stream>>>(W_ih_c, Wihcb, 4608 * 2304);
  cvt_f32_bf16_kernel<<<1024, 256, 0, stream>>>(W_hh_c, Whhcb, 4608 * 1536);
  cvt_f32_bf16_kernel<<<1024, 256, 0, stream>>>(W_ih_e, Wiheb, 4608 * 1536);
  cvt_f32_bf16_kernel<<<1024, 256, 0, stream>>>(W_hh_e, Whheb, 4608 * 1536);
  cvt_f32_bf16_kernel<<<1024, 256, 0, stream>>>(W_ih_t, Wihtb, 2304 * 1536);
  cvt_f32_bf16_kernel<<<512,  256, 0, stream>>>(W_hh_t, Whhtb, 2304 * 768);
  zero_u32_kernel<<<1, 64, 0, stream>>>(cnts, 64);

  // --- GRU c: input projection + recurrence ---
  wmma_gemm_xg<<<dim3(4608 / 128, MR / 128), 256, 0, stream>>>(opb, Wihcb, b_ih_c, xg_c, MR, 4608, 2304);
  zero_u32_kernel<<<48, 256, 0, stream>>>((unsigned*)hF, 8 * 1536);
  zero_u32_kernel<<<48, 256, 0, stream>>>((unsigned*)hB, 16 * 1536 / 2);
  gru_recur_kernel<<<96, 128, 0, stream>>>(Whhcb, xg_c, b_hh_c, rnn_c, rnn_cb, hF, hB, &cnts[0], 1536, 96);

  // --- GRU e & t input projections (consume bf16 rnn_c) ---
  wmma_gemm_xg<<<dim3(4608 / 128, MR / 128), 256, 0, stream>>>(rnn_cb, Wiheb, b_ih_e, xg_e, MR, 4608, 1536);
  wmma_gemm_xg<<<dim3(2304 / 128, MR / 128), 256, 0, stream>>>(rnn_cb, Wihtb, b_ih_t, xg_t, MR, 2304, 1536);

  // --- GRU e recurrence ---
  zero_u32_kernel<<<48, 256, 0, stream>>>((unsigned*)hF, 8 * 1536);
  zero_u32_kernel<<<48, 256, 0, stream>>>((unsigned*)hB, 16 * 1536 / 2);
  gru_recur_kernel<<<96, 128, 0, stream>>>(Whheb, xg_e, b_hh_e, rnn_e, nullptr, hF, hB, &cnts[1], 1536, 96);

  // --- GRU t recurrence ---
  zero_u32_kernel<<<24, 256, 0, stream>>>((unsigned*)hF, 8 * 768);
  zero_u32_kernel<<<24, 256, 0, stream>>>((unsigned*)hB, 16 * 768 / 2);
  gru_recur_kernel<<<48, 128, 0, stream>>>(Whhtb, xg_t, b_hh_t, rnn_t, nullptr, hF, hB, &cnts[2], 768, 48);

  // --- output heads ---
  fop1_kernel<<<MR, 256, 0, stream>>>(rnn_e, rnn_c, W1, b1, out);
  proj2_kernel<<<MR, 256, 0, stream>>>(rnn_t, W2, ABbuf);
  fop2_kernel<<<(MR * 10 + 255) / 256, 256, 0, stream>>>(ABbuf, b2, out + (size_t)MR * 7);
}
